// DeepseekV32Attention_87832081203521
// MI455X (gfx1250) — matrix operand order repaired and checked by reading, not compiled, not run on hardware
//
#include <hip/hip_runtime.h>
#include <math.h>

// ---------------------------------------------------------------------------
// Problem constants (match reference)
// ---------------------------------------------------------------------------
constexpr int S      = 2048;
constexpr int D      = 4096;
constexpr int H      = 16;
constexpr int DN     = 128;
constexpr int DR     = 64;
constexpr int DV     = 128;
constexpr int QLORA  = 1536;
constexpr int KVLORA = 512;
constexpr int IN_N   = 16;
constexpr int ID_D   = 128;
constexpr int TOPK   = 1024;

typedef __bf16 bf16;
typedef __attribute__((ext_vector_type(16))) __bf16 v16bf;
typedef __attribute__((ext_vector_type(8)))  __bf16 v8bf;
typedef __attribute__((ext_vector_type(8)))  float  v8f;

__device__ __forceinline__ bf16 to_bf16(float f) {
  unsigned u = __float_as_uint(f);
  u += 0x7fffu + ((u >> 16) & 1u);          // round-to-nearest-even
  unsigned short h = (unsigned short)(u >> 16);
  bf16 r;
  __builtin_memcpy(&r, &h, sizeof(h));
  return r;
}

__device__ __forceinline__ unsigned f2key(float f) {
  unsigned u = __float_as_uint(f);
  return (u & 0x80000000u) ? ~u : (u | 0x80000000u);   // monotone float->uint
}

// Fragment loaders: K is contiguous in memory for both operands (NT form).
// A 16-bit 16x32 layout: lanes 0-15 K={0..7,16..23}, lanes 16-31 K={8..15,24..31}
__device__ __forceinline__ v16bf ld_afrag(const bf16* __restrict__ row, int k0,
                                          int half) {
  v8bf lo = *(const v8bf*)(row + k0 + half * 8);
  v8bf hi = *(const v8bf*)(row + k0 + 16 + half * 8);
  return __builtin_shufflevector(lo, hi, 0, 1, 2, 3, 4, 5, 6, 7,
                                 8, 9, 10, 11, 12, 13, 14, 15);
}
// B 16-bit 32x16 layout: lane = column n, lanes 0-15 K={0..7,16..23}, lanes 16-31 K={8..15,24..31}
__device__ __forceinline__ v16bf ld_bfrag(const bf16* __restrict__ row, int k0,
                                          int half) {
  v8bf lo = *(const v8bf*)(row + k0 + half * 8);
  v8bf hi = *(const v8bf*)(row + k0 + 16 + half * 8);
  return __builtin_shufflevector(lo, hi, 0, 1, 2, 3, 4, 5, 6, 7,
                                 8, 9, 10, 11, 12, 13, 14, 15);
}

#define WMMA_BF16(a, b, c)                                                    \
  __builtin_amdgcn_wmma_f32_16x16x32_bf16(false, (a), false, (b), (short)0,   \
                                          (c), false, false)

// ---------------------------------------------------------------------------
// f32 -> bf16 convert (grid-stride)
// ---------------------------------------------------------------------------
__global__ __launch_bounds__(256)
void k_cvt(const float* __restrict__ src, bf16* __restrict__ dst, size_t n) {
  for (size_t i = (size_t)blockIdx.x * 256 + threadIdx.x; i < n;
       i += (size_t)gridDim.x * 256)
    dst[i] = to_bf16(src[i]);
}

// ---------------------------------------------------------------------------
// f32 -> bf16 transposing convert: dst[N][M] = src[M][N]^T  (32x32 LDS tiles)
// M, N multiples of 32 at all call sites.
// ---------------------------------------------------------------------------
__global__ __launch_bounds__(256)
void k_cvt_t(const float* __restrict__ src, int lds_, bf16* __restrict__ dst,
             int ldd) {
  __shared__ float tile[32][33];
  const int tid = threadIdx.x;
  const int rb = blockIdx.y * 32;     // source row base (M)
  const int cb = blockIdx.x * 32;     // source col base (N)
#pragma unroll
  for (int i = 0; i < 4; ++i) {
    int e = tid + i * 256, r = e >> 5, c = e & 31;
    tile[r][c] = src[(size_t)(rb + r) * lds_ + (cb + c)];
  }
  __syncthreads();
#pragma unroll
  for (int i = 0; i < 4; ++i) {
    int e = tid + i * 256, r = e >> 5, c = e & 31;   // r: n-local, c: m-local
    dst[(size_t)(cb + r) * ldd + (rb + c)] = to_bf16(tile[c][r]);
  }
}

// ---------------------------------------------------------------------------
// WMMA GEMM, NT form: C[M,N] (f32, ldc) = A[M,K] (bf16, lda) * Bt[N,K]^T (ldb)
// Direct global->fragment b128 loads, software-pipelined with UNCONDITIONAL
// next-step loads (one k-step over-read past K; operands are always interior
// workspace buffers and the workspace has a trailing guard region, so the
// over-read is mapped; garbage fragments are never consumed by a WMMA).
// Block = 256 threads = 8 waves arranged 4x2; wave tile 32x32 (4 WMMAs/step);
// block tile 128x64.
// Requirements (all call sites satisfy): M % 128 == 0, N % 64 == 0, K % 32 == 0,
// base pointers and (lda|ldb)*2 bytes 16B-aligned.
// ---------------------------------------------------------------------------
__global__ __launch_bounds__(256)
void k_gemm_nt(const bf16* __restrict__ A, int lda,
               const bf16* __restrict__ Bt, int ldb,
               float* __restrict__ C, int ldc,
               int K, int accumulate) {
  const int lane = threadIdx.x & 31;
  const int wave = threadIdx.x >> 5;
  const int wm   = wave >> 1;           // 0..3
  const int wn   = wave & 1;            // 0..1
  const int half = lane >> 4;           // K-half (A/B), M-half (C)
  const int mr   = lane & 15;
  const int tr = blockIdx.y * 128 + wm * 32;  // output row tile base
  const int tc = blockIdx.x * 64 + wn * 32;   // output col tile base

  const bf16* arow0 = A  + (size_t)(tr + mr) * lda;
  const bf16* arow1 = A  + (size_t)(tr + 16 + mr) * lda;
  const bf16* brow0 = Bt + (size_t)(tc + mr) * ldb;
  const bf16* brow1 = Bt + (size_t)(tc + 16 + mr) * ldb;

  v8f acc00 = {}, acc01 = {}, acc10 = {}, acc11 = {};

  // prologue: fragments for k0 = 0
  v16bf ca0 = ld_afrag(arow0, 0, half);
  v16bf ca1 = ld_afrag(arow1, 0, half);
  v16bf cb0 = ld_bfrag(brow0, 0, half);
  v16bf cb1 = ld_bfrag(brow1, 0, half);

#pragma unroll 2
  for (int k0 = 0; k0 < K; k0 += 32) {
    // unconditional next-step loads (over-read by one step on last iteration;
    // those fragments are never used)
    v16bf na0 = ld_afrag(arow0, k0 + 32, half);
    v16bf na1 = ld_afrag(arow1, k0 + 32, half);
    v16bf nb0 = ld_bfrag(brow0, k0 + 32, half);
    v16bf nb1 = ld_bfrag(brow1, k0 + 32, half);
    __builtin_prefetch(arow0 + k0 + 64, 0, 1);   // speculative, safe past end
    __builtin_prefetch(arow1 + k0 + 64, 0, 1);
    __builtin_prefetch(brow0 + k0 + 64, 0, 1);
    __builtin_prefetch(brow1 + k0 + 64, 0, 1);
    acc00 = WMMA_BF16(ca0, cb0, acc00);
    acc01 = WMMA_BF16(ca0, cb1, acc01);
    acc10 = WMMA_BF16(ca1, cb0, acc10);
    acc11 = WMMA_BF16(ca1, cb1, acc11);
    ca0 = na0; ca1 = na1; cb0 = nb0; cb1 = nb1;
  }

#pragma unroll
  for (int r = 0; r < 8; ++r) {                 // C/D: vgpr r -> M = r + 8*half
    int m0 = tr + r + 8 * half;
    int m1 = tr + 16 + r + 8 * half;
    size_t i00 = (size_t)m0 * ldc + tc + mr;
    size_t i01 = (size_t)m0 * ldc + tc + 16 + mr;
    size_t i10 = (size_t)m1 * ldc + tc + mr;
    size_t i11 = (size_t)m1 * ldc + tc + 16 + mr;
    C[i00] = acc00[r] + (accumulate ? C[i00] : 0.f);
    C[i01] = acc01[r] + (accumulate ? C[i01] : 0.f);
    C[i10] = acc10[r] + (accumulate ? C[i10] : 0.f);
    C[i11] = acc11[r] + (accumulate ? C[i11] : 0.f);
  }
}

// ---------------------------------------------------------------------------
// RMSNorm rows of x (f32, row stride xstride, first ncols cols) -> bf16
// ---------------------------------------------------------------------------
__global__ __launch_bounds__(256)
void k_rmsnorm(const float* __restrict__ x, int xstride,
               const float* __restrict__ w, bf16* __restrict__ y, int ncols) {
  int r = blockIdx.x;
  const float* row = x + (size_t)r * xstride;
  __shared__ float red[256];
  __shared__ float s_scale;
  float acc = 0.f;
  for (int c = threadIdx.x; c < ncols; c += 256) { float v = row[c]; acc += v * v; }
  red[threadIdx.x] = acc; __syncthreads();
  for (int off = 128; off > 0; off >>= 1) {
    if (threadIdx.x < off) red[threadIdx.x] += red[threadIdx.x + off];
    __syncthreads();
  }
  if (threadIdx.x == 0) s_scale = rsqrtf(red[0] / (float)ncols + 1e-6f);
  __syncthreads();
  float sc = s_scale;
  for (int c = threadIdx.x; c < ncols; c += 256)
    y[(size_t)r * ncols + c] = to_bf16(row[c] * sc * w[c]);
}

// ---------------------------------------------------------------------------
// Interleaved RoPE on q[..., DN:DN+DR] per head, fused with bf16 convert
// ---------------------------------------------------------------------------
__global__ __launch_bounds__(256)
void k_rope_q(const float* __restrict__ q, const float* __restrict__ cosb,
              const float* __restrict__ sinb, bf16* __restrict__ qbf) {
  int s = blockIdx.x;
  const int RW = H * (DN + DR);
  const float* row = q + (size_t)s * RW;
  bf16* orow = qbf + (size_t)s * RW;
  const float* cs = cosb + s * (DR / 2);
  const float* sn = sinb + s * (DR / 2);
  for (int d = threadIdx.x; d < RW; d += 256) {
    int r = d % (DN + DR);
    float v;
    if (r < DN) {
      v = row[d];
    } else {
      int idx = r - DN;                 // 0..63, interleaved pairs
      int j = idx >> 1;
      int base = d - (idx & 1);
      float xr = row[base], xi = row[base + 1];
      v = (idx & 1) ? (xr * sn[j] + xi * cs[j]) : (xr * cs[j] - xi * sn[j]);
    }
    orow[d] = to_bf16(v);
  }
}

// ---------------------------------------------------------------------------
// Interleaved RoPE on kv[..., KVLORA:KVLORA+DR] -> bf16 k_rope (S x DR)
// ---------------------------------------------------------------------------
__global__ __launch_bounds__(32)
void k_krope(const float* __restrict__ kv, const float* __restrict__ cosb,
             const float* __restrict__ sinb, bf16* __restrict__ krbf) {
  int s = blockIdx.x;
  int j = threadIdx.x;                  // 0..31
  float xr = kv[(size_t)s * (KVLORA + DR) + KVLORA + 2 * j];
  float xi = kv[(size_t)s * (KVLORA + DR) + KVLORA + 2 * j + 1];
  float c = cosb[s * 32 + j], sn = sinb[s * 32 + j];
  krbf[(size_t)s * DR + 2 * j]     = to_bf16(xr * c - xi * sn);
  krbf[(size_t)s * DR + 2 * j + 1] = to_bf16(xr * sn + xi * c);
}

// ---------------------------------------------------------------------------
// Non-interleaved RoPE on first DR dims of each 128-wide indexer query chunk
// ---------------------------------------------------------------------------
__global__ __launch_bounds__(256)
void k_rope_qi(const float* __restrict__ qi, const float* __restrict__ cosb,
               const float* __restrict__ sinb, bf16* __restrict__ out) {
  int s = blockIdx.x;
  const float* row = qi + (size_t)s * (IN_N * ID_D);
  bf16* orow = out + (size_t)s * (IN_N * ID_D);
  const float* cs = cosb + s * 32;
  const float* sn = sinb + s * 32;
  for (int d = threadIdx.x; d < IN_N * ID_D; d += 256) {
    int n = d >> 7, r = d & 127;
    const float* x = row + (n << 7);
    float v;
    if (r < 32)        v = x[r] * cs[r] - x[r + 32] * sn[r];
    else if (r < 64) { int j = r - 32; v = x[j] * sn[j] + x[r] * cs[j]; }
    else               v = x[r];
    orow[d] = to_bf16(v);
  }
}

// ---------------------------------------------------------------------------
// LayerNorm (128) + non-interleaved RoPE on first 64 dims -> bf16 ki
// ---------------------------------------------------------------------------
__global__ __launch_bounds__(128)
void k_ln_rope_ki(const float* __restrict__ ki, const float* __restrict__ w,
                  const float* __restrict__ b, const float* __restrict__ cosb,
                  const float* __restrict__ sinb, bf16* __restrict__ out) {
  int s = blockIdx.x;
  int c = threadIdx.x;                  // 0..127
  __shared__ float red[128];
  __shared__ float arr[128];
  __shared__ float s_mu, s_rs;
  float x = ki[(size_t)s * ID_D + c];
  red[c] = x; __syncthreads();
  for (int off = 64; off > 0; off >>= 1) {
    if (c < off) red[c] += red[c + off];
    __syncthreads();
  }
  if (c == 0) s_mu = red[0] / 128.f;
  __syncthreads();
  float d0 = x - s_mu;
  red[c] = d0 * d0; __syncthreads();
  for (int off = 64; off > 0; off >>= 1) {
    if (c < off) red[c] += red[c + off];
    __syncthreads();
  }
  if (c == 0) s_rs = rsqrtf(red[0] / 128.f + 1e-5f);
  __syncthreads();
  arr[c] = d0 * s_rs * w[c] + b[c];
  __syncthreads();
  const float* cs = cosb + s * 32;
  const float* sn = sinb + s * 32;
  float v;
  if (c < 32)        v = arr[c] * cs[c] - arr[c + 32] * sn[c];
  else if (c < 64) { int j = c - 32; v = arr[j] * sn[j] + arr[c] * cs[j]; }
  else               v = arr[c];
  out[(size_t)s * ID_D + c] = to_bf16(v);
}

// ---------------------------------------------------------------------------
// wts = hidden @ idx_w_proj * IN^-0.5     (N=16, done without WMMA)
// ---------------------------------------------------------------------------
__global__ __launch_bounds__(256)
void k_wts(const float* __restrict__ hidden, const float* __restrict__ w,
           float* __restrict__ wts) {
  int s = blockIdx.x;
  __shared__ float red[256];
  const float* hrow = hidden + (size_t)s * D;
  for (int n = 0; n < IN_N; ++n) {
    float acc = 0.f;
    for (int d = threadIdx.x; d < D; d += 256)
      acc += hrow[d] * w[(size_t)d * IN_N + n];
    red[threadIdx.x] = acc; __syncthreads();
    for (int off = 128; off > 0; off >>= 1) {
      if (threadIdx.x < off) red[threadIdx.x] += red[threadIdx.x + off];
      __syncthreads();
    }
    if (threadIdx.x == 0) wts[s * IN_N + n] = red[0] * 0.25f;  // 16^-0.5
    __syncthreads();
  }
}

// ---------------------------------------------------------------------------
// Indexer epilogue: iscores += relu(tmp * ID^-0.5) * wts[q][n]
// ---------------------------------------------------------------------------
__global__ __launch_bounds__(256)
void k_iscore_ep(const float* __restrict__ tmp, const float* __restrict__ wts,
                 float* __restrict__ isc, int n, int init) {
  const size_t total = (size_t)S * S;
  for (size_t i = (size_t)blockIdx.x * 256 + threadIdx.x; i < total;
       i += (size_t)gridDim.x * 256) {
    int q = (int)(i / (size_t)S);
    float v = tmp[i] * 0.08838834764831845f;   // 128^-0.5
    v = v > 0.f ? v : 0.f;
    float add = v * wts[q * IN_N + n];
    isc[i] = (init ? 0.f : isc[i]) + add;
  }
}

// ---------------------------------------------------------------------------
// Per-row radix top-k threshold over causal prefix k<=q.
// thr[q] = smallest selected key (monotone uint key); allowed iff key >= thr.
// ---------------------------------------------------------------------------
__global__ __launch_bounds__(256)
void k_topk(const float* __restrict__ isc, unsigned* __restrict__ thr) {
  int q = blockIdx.x;
  if (q + 1 <= TOPK) { if (threadIdx.x == 0) thr[q] = 0u; return; }
  const float* row = isc + (size_t)q * S;
  __shared__ int s_cnt;
  unsigned prefix = 0;
  int need = TOPK;
  for (int b = 31; b >= 0; --b) {
    if (threadIdx.x == 0) s_cnt = 0;
    __syncthreads();
    int local = 0;
    for (int k = threadIdx.x; k <= q; k += 256) {
      unsigned key = f2key(row[k]);
      unsigned hi = (b == 31) ? 0u : ((key ^ prefix) >> (b + 1));
      if (hi == 0u && ((key >> b) & 1u)) local++;
    }
    atomicAdd(&s_cnt, local);
    __syncthreads();
    int cnt = s_cnt;
    if (cnt >= need) prefix |= (1u << b);
    else need -= cnt;
    __syncthreads();
  }
  if (threadIdx.x == 0) thr[q] = prefix;
}

// ---------------------------------------------------------------------------
// Masked softmax of one head's scores -> bf16 probabilities
// ---------------------------------------------------------------------------
__global__ __launch_bounds__(256)
void k_softmax(const float* __restrict__ sc, const unsigned* __restrict__ thr,
               const float* __restrict__ isc, bf16* __restrict__ probs,
               float scale) {
  int q = blockIdx.x;
  const float* row  = sc  + (size_t)q * S;
  const float* irow = isc + (size_t)q * S;
  bf16* prow = probs + (size_t)q * S;
  unsigned t = thr[q];
  __shared__ float red[256];
  __shared__ float s_m, s_inv;
  float m = -3.0e38f;
  for (int k = threadIdx.x; k < S; k += 256)
    if (k <= q && f2key(irow[k]) >= t) m = fmaxf(m, row[k] * scale);
  red[threadIdx.x] = m; __syncthreads();
  for (int off = 128; off > 0; off >>= 1) {
    if (threadIdx.x < off)
      red[threadIdx.x] = fmaxf(red[threadIdx.x], red[threadIdx.x + off]);
    __syncthreads();
  }
  if (threadIdx.x == 0) s_m = red[0];
  __syncthreads();
  float mm = s_m;
  float sum = 0.f;
  for (int k = threadIdx.x; k < S; k += 256) {
    bool ok = (k <= q) && (f2key(irow[k]) >= t);
    sum += ok ? __expf(row[k] * scale - mm) : 0.f;
  }
  red[threadIdx.x] = sum; __syncthreads();
  for (int off = 128; off > 0; off >>= 1) {
    if (threadIdx.x < off) red[threadIdx.x] += red[threadIdx.x + off];
    __syncthreads();
  }
  if (threadIdx.x == 0) s_inv = 1.f / fmaxf(red[0], 1e-30f);
  __syncthreads();
  float inv = s_inv;
  for (int k = threadIdx.x; k < S; k += 256) {
    bool ok = (k <= q) && (f2key(irow[k]) >= t);
    float p = ok ? __expf(row[k] * scale - mm) * inv : 0.f;
    prow[k] = to_bf16(p);
  }
}

// ---------------------------------------------------------------------------
// Host orchestration
// ---------------------------------------------------------------------------
extern "C" void kernel_launch(void* const* d_in, const int* in_sizes, int n_in,
                              void* d_out, int out_size, void* d_ws, size_t ws_size,
                              hipStream_t stream) {
  (void)in_sizes; (void)n_in; (void)out_size; (void)ws_size;
  const float* hidden     = (const float*)d_in[0];
  const float* cosb       = (const float*)d_in[1];
  const float* sinb       = (const float*)d_in[2];
  const float* w_q_a      = (const float*)d_in[3];
  const float* q_a_ln_w   = (const float*)d_in[4];
  const float* w_q_b      = (const float*)d_in[5];
  const float* w_kv_a     = (const float*)d_in[6];
  const float* kv_a_ln_w  = (const float*)d_in[7];
  const float* w_kv_b     = (const float*)d_in[8];
  const float* w_o        = (const float*)d_in[9];
  const float* idx_wq_b   = (const float*)d_in[10];
  const float* idx_wk     = (const float*)d_in[11];
  const float* idx_k_ln_w = (const float*)d_in[12];
  const float* idx_k_ln_b = (const float*)d_in[13];
  const float* idx_w_proj = (const float*)d_in[14];
  float* out = (float*)d_out;

  // Bump allocator over workspace
  char* base = (char*)d_ws;
  size_t off = 0;
  auto alloc = [&](size_t bytes) -> void* {
    void* p = base + off;
    off = (off + bytes + 255) & ~(size_t)255;
    return p;
  };

  // bf16 buffers (weights stored TRANSPOSED: [N][K] so all GEMMs are NT form)
  bf16* hidden_bf = (bf16*)alloc((size_t)S * D * 2);
  bf16* wqaT      = (bf16*)alloc((size_t)QLORA * D * 2);                // [1536][4096]
  bf16* wqbT      = (bf16*)alloc((size_t)(H * (DN + DR)) * QLORA * 2);  // [3072][1536]
  bf16* wkvaT     = (bf16*)alloc((size_t)(KVLORA + DR) * D * 2);        // [576][4096]
  bf16* wkvbT     = (bf16*)alloc((size_t)(H * (DN + DV)) * KVLORA * 2); // [4096][512]
  bf16* woT       = (bf16*)alloc((size_t)D * (H * DV) * 2);             // [4096][2048]
  bf16* idxwqbT   = (bf16*)alloc((size_t)(IN_N * ID_D) * QLORA * 2);    // [2048][1536]
  bf16* idxwkT    = (bf16*)alloc((size_t)ID_D * D * 2);                 // [128][4096]
  bf16* cq_bf     = (bf16*)alloc((size_t)S * QLORA * 2);
  bf16* q_bf      = (bf16*)alloc((size_t)S * H * (DN + DR) * 2);
  bf16* kvc_bf    = (bf16*)alloc((size_t)S * KVLORA * 2);
  bf16* kvb_bf    = (bf16*)alloc((size_t)S * H * (DN + DV) * 2);   // token-major
  bf16* kvbT      = (bf16*)alloc((size_t)(H * (DN + DV)) * S * 2); // dim-major
  bf16* kr_bf     = (bf16*)alloc((size_t)S * DR * 2);
  bf16* qi_bf     = (bf16*)alloc((size_t)S * IN_N * ID_D * 2);
  bf16* ki_bf     = (bf16*)alloc((size_t)S * ID_D * 2);
  bf16* probs_bf  = (bf16*)alloc((size_t)S * S * 2);
  bf16* attn_bf   = (bf16*)alloc((size_t)S * H * DV * 2);
  // f32 buffers
  float*    ftmp    = (float*)alloc((size_t)S * 4096 * 4);   // recycled GEMM output
  float*    iscores = (float*)alloc((size_t)S * S * 4);
  float*    attn_f  = (float*)alloc((size_t)S * H * DV * 4);
  float*    wts_f   = (float*)alloc((size_t)S * IN_N * 4);
  unsigned* thr     = (unsigned*)alloc((size_t)S * 4);
  (void)alloc(4096);                       // guard region for GEMM over-reads

  // gemm: C[M,N] = A[M,K] * Bt[N,K]^T
  auto gemm = [&](const bf16* A, int lda, const bf16* Bt, int ldb,
                  float* C, int ldc, int M, int N, int K, int acc) {
    dim3 g(N / 64, M / 128);
    k_gemm_nt<<<g, 256, 0, stream>>>(A, lda, Bt, ldb, C, ldc, K, acc);
  };
  // transposing convert: dst[n][m] = src[m][n], src is Mrows x Ncols
  auto cvt_t = [&](const float* src, int Mrows, int Ncols, bf16* dst) {
    dim3 g(Ncols / 32, Mrows / 32);
    k_cvt_t<<<g, 256, 0, stream>>>(src, Ncols, dst, Mrows);
  };

  // --- bf16 conversions (weights transposed) ---
  k_cvt<<<4096, 256, 0, stream>>>(hidden, hidden_bf, (size_t)S * D);
  cvt_t(w_q_a,    D,      QLORA,           wqaT);
  cvt_t(w_q_b,    QLORA,  H * (DN + DR),   wqbT);
  cvt_t(w_kv_a,   D,      KVLORA + DR,     wkvaT);
  cvt_t(w_kv_b,   KVLORA, H * (DN + DV),   wkvbT);
  cvt_t(w_o,      H * DV, D,               woT);
  cvt_t(idx_wq_b, QLORA,  IN_N * ID_D,     idxwqbT);
  cvt_t(idx_wk,   D,      ID_D,            idxwkT);

  // --- cq = rmsnorm(hidden @ w_q_a) ---
  gemm(hidden_bf, D, wqaT, D, ftmp, QLORA, S, QLORA, D, 0);
  k_rmsnorm<<<S, 256, 0, stream>>>(ftmp, QLORA, q_a_ln_w, cq_bf, QLORA);

  // --- q = rope(cq @ w_q_b) ---
  gemm(cq_bf, QLORA, wqbT, QLORA, ftmp, H * (DN + DR), S, H * (DN + DR), QLORA, 0);
  k_rope_q<<<S, 256, 0, stream>>>(ftmp, cosb, sinb, q_bf);

  // --- kv = hidden @ w_kv_a; kv_c = rmsnorm; k_rope ---
  gemm(hidden_bf, D, wkvaT, D, ftmp, KVLORA + DR, S, KVLORA + DR, D, 0);
  k_rmsnorm<<<S, 256, 0, stream>>>(ftmp, KVLORA + DR, kv_a_ln_w, kvc_bf, KVLORA);
  k_krope<<<S, 32, 0, stream>>>(ftmp, cosb, sinb, kr_bf);

  // --- kvb = kv_c @ w_kv_b  (keep both orientations) ---
  gemm(kvc_bf, KVLORA, wkvbT, KVLORA, ftmp, H * (DN + DV), S, H * (DN + DV), KVLORA, 0);
  k_cvt<<<4096, 256, 0, stream>>>(ftmp, kvb_bf, (size_t)S * H * (DN + DV));
  cvt_t(ftmp, S, H * (DN + DV), kvbT);

  // --- qi = rope_noninterleaved(cq @ idx_wq_b) ---
  gemm(cq_bf, QLORA, idxwqbT, QLORA, ftmp, IN_N * ID_D, S, IN_N * ID_D, QLORA, 0);
  k_rope_qi<<<S, 256, 0, stream>>>(ftmp, cosb, sinb, qi_bf);

  // --- ki = rope(layernorm(hidden @ idx_wk)) ---
  gemm(hidden_bf, D, idxwkT, D, ftmp, ID_D, S, ID_D, D, 0);
  k_ln_rope_ki<<<S, 128, 0, stream>>>(ftmp, idx_k_ln_w, idx_k_ln_b,
                                      cosb, sinb, ki_bf);

  // --- wts ---
  k_wts<<<S, 256, 0, stream>>>(hidden, idx_w_proj, wts_f);

  // --- indexer scores: sum_n relu(qi_n @ ki^T * ID^-0.5) * wts[:,n] ---
  for (int n = 0; n < IN_N; ++n) {
    gemm(qi_bf + n * ID_D, IN_N * ID_D, ki_bf, ID_D, ftmp, S, S, S, ID_D, 0);
    k_iscore_ep<<<4096, 256, 0, stream>>>(ftmp, wts_f, iscores, n, n == 0);
  }

  // --- top-k thresholds ---
  k_topk<<<S, 256, 0, stream>>>(iscores, thr);

  // --- attention per head ---
  const float scale = 0.07216878364870323f;    // (DN+DR)^-0.5
  for (int h = 0; h < H; ++h) {
    gemm(q_bf + h * (DN + DR), H * (DN + DR),
         kvb_bf + h * (DN + DV), H * (DN + DV),
         ftmp, S, S, S, DN, 0);                          // nope scores
    gemm(q_bf + h * (DN + DR) + DN, H * (DN + DR),
         kr_bf, DR, ftmp, S, S, S, DR, 1);               // + rope scores
    k_softmax<<<S, 256, 0, stream>>>(ftmp, thr, iscores, probs_bf, scale);
    gemm(probs_bf, S, kvbT + (size_t)(h * (DN + DV) + DN) * S, S,
         attn_f + h * DV, H * DV, S, DV, S, 0);          // P @ V (dim-major V)
  }

  // --- out = attn @ w_o ---
  k_cvt<<<4096, 256, 0, stream>>>(attn_f, attn_bf, (size_t)S * H * DV);
  gemm(attn_bf, H * DV, woT, H * DV, out, D, S, D, H * DV, 0);
}
